// HGGNet_33079838114101
// MI455X (gfx1250) — matrix operand order repaired and checked by reading, not compiled, not run on hardware
//
#include <hip/hip_runtime.h>
#include <math.h>

// ---------------------------------------------------------------------------
// HGGNet forward for MI455X (gfx1250, wave32, WMMA).
// Workhorse: f16 WMMA 16x16x32 GEMM, f32 accumulate, 32x32 macro-tile/wave
// (4 independent WMMAs/k-step), B always pre-transposed -> contiguous-K
// b128 fragment loads, branch-free bounds handling (clamp+select).
// ---------------------------------------------------------------------------

typedef __attribute__((ext_vector_type(16))) _Float16 v16h;
typedef __attribute__((ext_vector_type(8)))  float    v8f;

#define BATCH  8
#define NWAVES 8   // waves per 256-thread block

__device__ __forceinline__ float hg_act(float v, int act) {
  if (act == 1) return v > 0.f ? v : 0.f;          // relu
  if (act == 2) return v > 0.f ? v : 0.2f * v;     // leaky 0.2
  if (act == 3) return 1.f / (1.f + __expf(-v));   // sigmoid
  return v;
}

__device__ __forceinline__ v8f wmma32(v16h a, v16h b, v8f c) {
  return __builtin_amdgcn_wmma_f32_16x16x32_f16(false, a, false, b, (short)0, c, false, false);
}

// ---- fragment loaders -----------------------------------------------------
// A (16x32 f16) lane layout: afr[j]   = arow[k0 + half*16 + j]      (j=0..7)
//                            afr[8+j] = arow[k0 + half*16 + 8 + j]
// B^T row layout (B stored [N][K]): bfr[j] = bcol[k0 + half*16 + j] (j=0..15)

__device__ __forceinline__ v16h afrag_vec(const float* __restrict__ arow,
                                          int k0, int half) {
  const float4 a0 = *(const float4*)(arow + k0 + half * 16);
  const float4 a1 = *(const float4*)(arow + k0 + half * 16 + 4);
  const float4 a2 = *(const float4*)(arow + k0 + half * 16 + 8);
  const float4 a3 = *(const float4*)(arow + k0 + half * 16 + 12);
  v16h f;
  f[0]=(_Float16)a0.x; f[1]=(_Float16)a0.y; f[2]=(_Float16)a0.z; f[3]=(_Float16)a0.w;
  f[4]=(_Float16)a1.x; f[5]=(_Float16)a1.y; f[6]=(_Float16)a1.z; f[7]=(_Float16)a1.w;
  f[8]=(_Float16)a2.x; f[9]=(_Float16)a2.y; f[10]=(_Float16)a2.z; f[11]=(_Float16)a2.w;
  f[12]=(_Float16)a3.x; f[13]=(_Float16)a3.y; f[14]=(_Float16)a3.z; f[15]=(_Float16)a3.w;
  return f;
}

__device__ __forceinline__ v16h afrag_sc(const float* __restrict__ arow,
                                         int k0, int half) {
  v16h f;
#pragma unroll
  for (int j = 0; j < 8; ++j) {
    f[j]     = (_Float16)arow[k0 + half * 16 + j];
    f[8 + j] = (_Float16)arow[k0 + half * 16 + 8 + j];
  }
  return f;
}

__device__ __forceinline__ v16h afrag_tail(const float* __restrict__ arow,
                                           int k0, int half, int Kd) {
  v16h f;
#pragma unroll
  for (int j = 0; j < 8; ++j) {
    int k1 = k0 + half * 16 + j;
    int k2 = k0 + half * 16 + 8 + j;
    float x1 = arow[k1 < Kd ? k1 : Kd - 1];
    float x2 = arow[k2 < Kd ? k2 : Kd - 1];
    f[j]     = (_Float16)(k1 < Kd ? x1 : 0.f);
    f[8 + j] = (_Float16)(k2 < Kd ? x2 : 0.f);
  }
  return f;
}

__device__ __forceinline__ v16h bfrag_vec(const float* __restrict__ bcol, int kbase) {
  const float4 b0 = *(const float4*)(bcol + kbase);
  const float4 b1 = *(const float4*)(bcol + kbase + 4);
  const float4 b2 = *(const float4*)(bcol + kbase + 8);
  const float4 b3 = *(const float4*)(bcol + kbase + 12);
  v16h f;
  f[0]=(_Float16)b0.x; f[1]=(_Float16)b0.y; f[2]=(_Float16)b0.z; f[3]=(_Float16)b0.w;
  f[4]=(_Float16)b1.x; f[5]=(_Float16)b1.y; f[6]=(_Float16)b1.z; f[7]=(_Float16)b1.w;
  f[8]=(_Float16)b2.x; f[9]=(_Float16)b2.y; f[10]=(_Float16)b2.z; f[11]=(_Float16)b2.w;
  f[12]=(_Float16)b3.x; f[13]=(_Float16)b3.y; f[14]=(_Float16)b3.z; f[15]=(_Float16)b3.w;
  return f;
}

__device__ __forceinline__ v16h bfrag_sc(const float* __restrict__ bcol, int kbase) {
  v16h f;
#pragma unroll
  for (int j = 0; j < 16; ++j) f[j] = (_Float16)bcol[kbase + j];
  return f;
}

__device__ __forceinline__ v16h bfrag_tail(const float* __restrict__ bcol,
                                           int kbase, int Kd) {
  v16h f;
#pragma unroll
  for (int j = 0; j < 16; ++j) {
    int k = kbase + j;
    float x = bcol[k < Kd ? k : Kd - 1];
    f[j] = (_Float16)(k < Kd ? x : 0.f);
  }
  return f;
}

// fully guarded 16x16 tile (edge tiles); clamp addresses, select values.
// B is stored transposed: element (k, col) at Bb[col*ldb + k].
__device__ v8f gemm_tile_slow(const float* __restrict__ Ab, const float* __restrict__ Bb,
                              int M, int N, int Kd, int lda, int ldb,
                              int tm, int tn, int half, int l16)
{
  const int row = tm * 16 + l16;
  const int col = tn * 16 + l16;
  const int rc = row < M ? row : M - 1;
  const int cc = col < N ? col : N - 1;
  const int rok = row < M, cok = col < N;
  v8f acc = {};
  for (int k0 = 0; k0 < Kd; k0 += 32) {
    v16h afr, bfr;
#pragma unroll
    for (int j = 0; j < 8; ++j) {
      int k1 = k0 + half * 16 + j;
      int k2 = k0 + half * 16 + 8 + j;
      float x1 = Ab[(long)rc * lda + (k1 < Kd ? k1 : Kd - 1)];
      float x2 = Ab[(long)rc * lda + (k2 < Kd ? k2 : Kd - 1)];
      afr[j]     = (_Float16)((rok && k1 < Kd) ? x1 : 0.f);
      afr[8 + j] = (_Float16)((rok && k2 < Kd) ? x2 : 0.f);
    }
#pragma unroll
    for (int j = 0; j < 16; ++j) {
      int kk = k0 + half * 16 + j;
      float x = Bb[(long)cc * ldb + (kk < Kd ? kk : Kd - 1)];
      bfr[j] = (_Float16)((cok && kk < Kd) ? x : 0.f);
    }
    acc = wmma32(afr, bfr, acc);
  }
  return acc;
}

__device__ __forceinline__ void store_tile(float* __restrict__ Cb, v8f acc, int tm, int tn,
                                           int half, int l16, int M, int N, int ldc,
                                           const float* __restrict__ bias, int act,
                                           float alpha)
{
  const int col = tn * 16 + l16;
#pragma unroll
  for (int r = 0; r < 8; ++r) {
    int cm = tm * 16 + r + half * 8;
    if (cm < M && col < N) {
      float v = acc[r] * alpha;
      if (bias) v += bias[col];
      Cb[(long)cm * ldc + col] = hg_act(v, act);
    }
  }
}

// ---------------------------------------------------------------------------
// WMMA GEMM:  C[z] = act( alpha * A[z] @ B[z]^T' (+bias) )
// B is ALWAYS stored transposed ([N][K], ld=ldb): element (k,n) = B[n*ldb+k].
// One wave computes a 32x32 C macro-tile: 2 A frags x 2 B frags ->
// 4 independent WMMAs per 32-deep k-step. Grid.x covers
// ceil(M/32)*ceil(N/32) macro-tiles across NWAVES waves/block.
// Two-level batch: z = z1*B2 + z2 with separate strides (used for [b,h]).
// ---------------------------------------------------------------------------
__global__ void k_gemm_wmma(const float* __restrict__ A, const float* __restrict__ Bw,
                            const float* __restrict__ bias, float* __restrict__ C,
                            int M, int N, int Kd, int lda, int ldb, int ldc,
                            int act, float alpha, int B2,
                            long sA1, long sA2, long sB1, long sB2, long sC1, long sC2)
{
  const int z  = blockIdx.y;
  const int z1 = z / B2, z2 = z % B2;
  const float* Ab = A  + (long)z1 * sA1 + (long)z2 * sA2;
  const float* Bb = Bw + (long)z1 * sB1 + (long)z2 * sB2;
  float*       Cb = C  + (long)z1 * sC1 + (long)z2 * sC2;

  const int lane    = threadIdx.x & 31;
  const int wv      = threadIdx.x >> 5;
  const int tilesN2 = (N + 31) >> 5;
  const int tilesM2 = (M + 31) >> 5;
  const int tile    = blockIdx.x * NWAVES + wv;
  if (tile >= tilesM2 * tilesN2) return;       // uniform per wave
  const int tm2  = tile / tilesN2;
  const int tn2  = tile % tilesN2;
  const int half = lane >> 4;
  const int l16  = lane & 15;

  if (tm2 * 32 + 32 <= M && tn2 * 32 + 32 <= N) {
    // ---------------- interior fast path: no guards ----------------
    const float* arow0 = Ab + (long)(tm2 * 32 + l16) * lda;
    const float* arow1 = arow0 + (long)16 * lda;
    const float* bcol0 = Bb + (long)(tn2 * 32 + l16) * ldb;
    const float* bcol1 = bcol0 + (long)16 * ldb;
    v8f acc00 = {}, acc01 = {}, acc10 = {}, acc11 = {};
    const int kmain = Kd & ~31;
    if (((lda | ldb) & 3) == 0) {
#pragma unroll 2
      for (int k0 = 0; k0 < kmain; k0 += 32) {
        v16h a0 = afrag_vec(arow0, k0, half);
        v16h a1 = afrag_vec(arow1, k0, half);
        v16h b0 = bfrag_vec(bcol0, k0 + half * 16);
        v16h b1 = bfrag_vec(bcol1, k0 + half * 16);
        acc00 = wmma32(a0, b0, acc00);
        acc01 = wmma32(a0, b1, acc01);
        acc10 = wmma32(a1, b0, acc10);
        acc11 = wmma32(a1, b1, acc11);
      }
    } else {
      for (int k0 = 0; k0 < kmain; k0 += 32) {
        v16h a0 = afrag_sc(arow0, k0, half);
        v16h a1 = afrag_sc(arow1, k0, half);
        v16h b0 = bfrag_sc(bcol0, k0 + half * 16);
        v16h b1 = bfrag_sc(bcol1, k0 + half * 16);
        acc00 = wmma32(a0, b0, acc00);
        acc01 = wmma32(a0, b1, acc01);
        acc10 = wmma32(a1, b0, acc10);
        acc11 = wmma32(a1, b1, acc11);
      }
    }
    if (kmain < Kd) {
      v16h a0 = afrag_tail(arow0, kmain, half, Kd);
      v16h a1 = afrag_tail(arow1, kmain, half, Kd);
      v16h b0 = bfrag_tail(bcol0, kmain + half * 16, Kd);
      v16h b1 = bfrag_tail(bcol1, kmain + half * 16, Kd);
      acc00 = wmma32(a0, b0, acc00);
      acc01 = wmma32(a0, b1, acc01);
      acc10 = wmma32(a1, b0, acc10);
      acc11 = wmma32(a1, b1, acc11);
    }
    store_tile(Cb, acc00, tm2 * 2,     tn2 * 2,     half, l16, M, N, ldc, bias, act, alpha);
    store_tile(Cb, acc01, tm2 * 2,     tn2 * 2 + 1, half, l16, M, N, ldc, bias, act, alpha);
    store_tile(Cb, acc10, tm2 * 2 + 1, tn2 * 2,     half, l16, M, N, ldc, bias, act, alpha);
    store_tile(Cb, acc11, tm2 * 2 + 1, tn2 * 2 + 1, half, l16, M, N, ldc, bias, act, alpha);
  } else {
    // ---------------- edge path: clamp+select, per 16x16 subtile -----------
    for (int sm = 0; sm < 2; ++sm) {
      int tm = tm2 * 2 + sm;
      if (tm * 16 >= M) break;
      for (int sn = 0; sn < 2; ++sn) {
        int tn = tn2 * 2 + sn;
        if (tn * 16 >= N) break;
        v8f acc = gemm_tile_slow(Ab, Bb, M, N, Kd, lda, ldb, tm, tn, half, l16);
        store_tile(Cb, acc, tm, tn, half, l16, M, N, ldc, bias, act, alpha);
      }
    }
  }
}

// --------------------------- helper kernels --------------------------------

// batched transpose: src [b][R][C] -> dst [b][C][R]
__global__ void k_transpose_b(const float* __restrict__ src, float* __restrict__ dst,
                              int R, int Cc)
{
  int b = blockIdx.y;
  long i = (long)blockIdx.x * blockDim.x + threadIdx.x;
  if (i >= (long)R * Cc) return;
  int r = (int)(i / Cc), c = (int)(i % Cc);
  dst[(long)b * R * Cc + (long)c * R + r] = src[(long)b * R * Cc + i];
}

__global__ void k_add(const float* a, const float* b, float* o, long n) {
  long i = (long)blockIdx.x * blockDim.x + threadIdx.x;
  if (i < n) o[i] = a[i] + b[i];
}

// FPS: one block per batch (matches jax scan: emit far, update dists, argmax)
__global__ void k_fps(const float* __restrict__ xyz, int* __restrict__ out,
                      int Npts, int nsample)
{
  __shared__ float dist[2048];
  __shared__ float rmax[256];
  __shared__ int   rid[256];
  const int b = blockIdx.x;
  const float* xb = xyz + (long)b * Npts * 3;
  const int t = threadIdx.x;
  for (int i = t; i < Npts; i += blockDim.x) dist[i] = 1e10f;
  __syncthreads();
  int far = 0;
  for (int s = 0; s < nsample; ++s) {
    if (t == 0) out[(long)b * nsample + s] = far;
    float fx = xb[far * 3], fy = xb[far * 3 + 1], fz = xb[far * 3 + 2];
    float best = -1.f; int bi = 0;
    for (int i = t; i < Npts; i += blockDim.x) {
      float dx = xb[i * 3] - fx, dy = xb[i * 3 + 1] - fy, dz = xb[i * 3 + 2] - fz;
      float m = fminf(dist[i], dx * dx + dy * dy + dz * dz);
      dist[i] = m;
      if (m > best) { best = m; bi = i; }
    }
    rmax[t] = best; rid[t] = bi;
    __syncthreads();
    for (int off = 128; off > 0; off >>= 1) {
      if (t < off) {
        if (rmax[t + off] > rmax[t] ||
            (rmax[t + off] == rmax[t] && rid[t + off] < rid[t])) {
          rmax[t] = rmax[t + off]; rid[t] = rid[t + off];
        }
      }
      __syncthreads();
    }
    far = rid[0];
    __syncthreads();
  }
}

__global__ void k_gather(const float* __restrict__ src, const int* __restrict__ idx,
                         float* __restrict__ dst, int Nsrc, int Mdst, int C)
{
  long i = (long)blockIdx.x * blockDim.x + threadIdx.x;
  int b = blockIdx.y;
  if (i >= (long)Mdst * C) return;
  int m = (int)(i / C), c = (int)(i % C);
  dst[((long)b * Mdst + m) * C + c] =
      src[((long)b * Nsrc + idx[(long)b * Mdst + m]) * C + c];
}

// brute-force K-smallest-distance (K<=16), insertion sort (ascending dist)
__global__ void k_knn(const float* __restrict__ q, const float* __restrict__ ref,
                      int* __restrict__ oidx, float* __restrict__ odist,
                      int Mq, int Nref, int Kk)
{
  long i = (long)blockIdx.x * blockDim.x + threadIdx.x;
  int b = blockIdx.y;
  if (i >= Mq) return;
  const float* qp = q + ((long)b * Mq + i) * 3;
  const float qx = qp[0], qy = qp[1], qz = qp[2];
  const float* rb = ref + (long)b * Nref * 3;
  float bd[16]; int bi[16];
  for (int k = 0; k < Kk; ++k) { bd[k] = 1e30f; bi[k] = 0; }
  for (int j = 0; j < Nref; ++j) {
    float dx = rb[j * 3] - qx, dy = rb[j * 3 + 1] - qy, dz = rb[j * 3 + 2] - qz;
    float d = dx * dx + dy * dy + dz * dz;
    if (d < bd[Kk - 1]) {
      int p = Kk - 1;
      while (p > 0 && bd[p - 1] > d) { bd[p] = bd[p - 1]; bi[p] = bi[p - 1]; --p; }
      bd[p] = d; bi[p] = j;
    }
  }
  for (int k = 0; k < Kk; ++k) {
    oidx[((long)b * Mq + i) * Kk + k] = bi[k];
    if (odist) odist[((long)b * Mq + i) * Kk + k] = bd[k];
  }
}

// e[b,m,k,0:C]=qf ; e[b,m,k,C:2C]=feat[nbr]-qf
__global__ void k_edge_feat(const float* __restrict__ qf, const float* __restrict__ feat,
                            const int* __restrict__ nbr, float* __restrict__ e,
                            int Mq, int Nref, int C, int Kk)
{
  long i = (long)blockIdx.x * blockDim.x + threadIdx.x;
  int b = blockIdx.y;
  if (i >= (long)Mq * Kk * C) return;
  int c = (int)(i % C); long mk = i / C;
  int k = (int)(mk % Kk); int m = (int)(mk / Kk);
  float qv = qf[((long)b * Mq + m) * C + c];
  int j = nbr[((long)b * Mq + m) * Kk + k];
  float gv = feat[((long)b * Nref + j) * C + c];
  float* row = e + (((long)b * Mq + m) * Kk + k) * (long)(2 * C);
  row[c]     = qv;
  row[C + c] = gv - qv;
}

__global__ void k_maxk(const float* __restrict__ e, float* __restrict__ out,
                       int Mq, int Kk, int C)
{
  long i = (long)blockIdx.x * blockDim.x + threadIdx.x;
  int b = blockIdx.y;
  if (i >= (long)Mq * C) return;
  int m = (int)(i / C), c = (int)(i % C);
  const float* base = e + (((long)b * Mq + m) * Kk) * (long)C + c;
  float v = base[0];
  for (int k = 1; k < Kk; ++k) v = fmaxf(v, base[(long)k * C]);
  out[((long)b * Mq + m) * C + c] = v;
}

__global__ void k_colmax(const float* in, float* out, int N, int C) {
  int b = blockIdx.y;
  int i = blockIdx.x * blockDim.x + threadIdx.x;
  if (i >= C) return;
  float v = -1e30f;
  for (int n = 0; n < N; ++n) v = fmaxf(v, in[((long)b * N + n) * C + i]);
  out[(long)b * C + i] = v;
}

__global__ void k_colmean2(const float* a, const float* bt, float* out, int N, int C) {
  int b = blockIdx.y;
  int i = blockIdx.x * blockDim.x + threadIdx.x;
  if (i >= C) return;
  float s = 0.f;
  for (int n = 0; n < N; ++n)
    s += a[((long)b * N + n) * C + i] + bt[((long)b * N + n) * C + i];
  out[(long)b * C + i] = s / (float)N;
}

__global__ void k_badd(const float* feat, const float* g, float* out, int N, int C) {
  long i = (long)blockIdx.x * blockDim.x + threadIdx.x;
  int b = blockIdx.y;
  if (i >= (long)N * C) return;
  int c = (int)(i % C);
  out[(long)b * N * C + i] = feat[(long)b * N * C + i] + g[(long)b * C + c];
}

__global__ void k_blend(const float* g, const float* a, const float* bt,
                        float* out, int N, int C) {
  long i = (long)blockIdx.x * blockDim.x + threadIdx.x;
  int b = blockIdx.y;
  if (i >= (long)N * C) return;
  int c = (int)(i % C);
  float gg = g[(long)b * C + c];
  out[(long)b * N * C + i] =
      gg * a[(long)b * N * C + i] + (1.f - gg) * bt[(long)b * N * C + i];
}

__global__ void k_softmax(float* x, int L) {
  __shared__ float red[256];
  float* r = x + (long)blockIdx.x * L;
  int t = threadIdx.x;
  float mx = -1e30f;
  for (int i = t; i < L; i += blockDim.x) mx = fmaxf(mx, r[i]);
  red[t] = mx; __syncthreads();
  for (int o = 128; o > 0; o >>= 1) { if (t < o) red[t] = fmaxf(red[t], red[t + o]); __syncthreads(); }
  mx = red[0]; __syncthreads();
  float s = 0.f;
  for (int i = t; i < L; i += blockDim.x) { float v = __expf(r[i] - mx); r[i] = v; s += v; }
  red[t] = s; __syncthreads();
  for (int o = 128; o > 0; o >>= 1) { if (t < o) red[t] += red[t + o]; __syncthreads(); }
  float inv = 1.f / red[0];
  for (int i = t; i < L; i += blockDim.x) r[i] *= inv;
}

__global__ void k_copy_cols(const float* src, int ld_src, float* dst, int ld_dst,
                            int col_off, long rows, int C)
{
  long i = (long)blockIdx.x * blockDim.x + threadIdx.x;
  if (i >= rows * (long)C) return;
  long r = i / C; int c = (int)(i % C);
  dst[r * (long)ld_dst + col_off + c] = src[r * (long)ld_src + c];
}

__global__ void k_interp3(const float* __restrict__ f2, const int* __restrict__ idx,
                          const float* __restrict__ dist, float* __restrict__ dst,
                          int ld_dst, int col_off, int M, int Nsrc, int C)
{
  long i = (long)blockIdx.x * blockDim.x + threadIdx.x;
  int b = blockIdx.y;
  if (i >= (long)M * C) return;
  int m = (int)(i / C), c = (int)(i % C);
  const float* dd = dist + ((long)b * M + m) * 3;
  const int*   ii = idx  + ((long)b * M + m) * 3;
  float w0 = 1.f / fmaxf(dd[0], 1e-10f);
  float w1 = 1.f / fmaxf(dd[1], 1e-10f);
  float w2 = 1.f / fmaxf(dd[2], 1e-10f);
  float inv = 1.f / (w0 + w1 + w2);
  float v = (w0 * f2[((long)b * Nsrc + ii[0]) * C + c] +
             w1 * f2[((long)b * Nsrc + ii[1]) * C + c] +
             w2 * f2[((long)b * Nsrc + ii[2]) * C + c]) * inv;
  dst[((long)b * M + m) * (long)ld_dst + col_off + c] = v;
}

__global__ void k_bcast_glob(const float* glob, float* dst, int ld, int off, int n, int C) {
  long i = (long)blockIdx.x * blockDim.x + threadIdx.x;
  int b = blockIdx.y;
  if (i >= (long)n * C) return;
  int m = (int)(i / C), c = (int)(i % C);
  dst[((long)b * n + m) * (long)ld + off + c] = glob[(long)b * C + c];
}

// center[b,m,c] = raw[b, c*1024 + m]; also mirrored into d_out region 0
__global__ void k_center(const float* raw, float* center, float* outp, int Mc) {
  long i = (long)blockIdx.x * blockDim.x + threadIdx.x;
  int b = blockIdx.y;
  if (i >= (long)Mc * 3) return;
  int m = (int)(i / 3), c = (int)(i % 3);
  float v = raw[(long)b * 3 * Mc + (long)c * Mc + m];
  center[((long)b * Mc + m) * 3 + c] = v;
  outp  [((long)b * Mc + m) * 3 + c] = v;
}

// fold input 1: [Mtot*16, 130] = [x(128) | grid(2)], grid = meshgrid ij of linspace(-1,1,4)
__global__ void k_fold_in1(const float* x, float* dst, long Mtot) {
  long i = (long)blockIdx.x * blockDim.x + threadIdx.x;
  if (i >= Mtot * 16 * 130) return;
  int c = (int)(i % 130); long mg = i / 130; int g = (int)(mg % 16); long m = mg / 16;
  float v;
  if (c < 128) v = x[m * 128 + c];
  else {
    int gi = (c == 128) ? (g / 4) : (g % 4);
    v = -1.f + (2.f / 3.f) * (float)gi;
  }
  dst[mg * 130 + c] = v;
}

// fold input 2: [Mtot*16, 131] = [x(128) | fd1(3)]
__global__ void k_fold_in2(const float* x, const float* fd1, float* dst, long Mtot) {
  long i = (long)blockIdx.x * blockDim.x + threadIdx.x;
  if (i >= Mtot * 16 * 131) return;
  int c = (int)(i % 131); long mg = i / 131; long m = mg / 16;
  dst[mg * 131 + c] = (c < 128) ? x[m * 128 + c] : fd1[mg * 3 + (c - 128)];
}

// out[b, ooff + (m*16+g)*3 + c] = fd2 + center point
__global__ void k_fold_out(const float* fd2, const float* cpts, float* outp,
                           int n, long ostride, long ooff)
{
  long i = (long)blockIdx.x * blockDim.x + threadIdx.x;
  int b = blockIdx.y;
  if (i >= (long)n * 48) return;
  int c = (int)(i % 3); long mg = i / 3; int g = (int)(mg % 16); int m = (int)(mg / 16);
  outp[(long)b * ostride + ooff + ((long)m * 16 + g) * 3 + c] =
      fd2[(((long)b * n + m) * 16 + g) * 3 + c] + cpts[((long)b * n + m) * 3 + c];
}

// --------------------------- host-side helpers -----------------------------

struct Lin { const float* w; const float* b; };
struct DT  { Lin pos, guide, q, k, v, proj; };

struct Arena { char* base; size_t off; size_t cap; };
static float* af(Arena& a, size_t n) {
  float* p = (float*)(a.base + a.off);
  a.off += ((n * sizeof(float) + 255) & ~(size_t)255);
  return p;
}
static int* ai(Arena& a, size_t n) {
  int* p = (int*)(a.base + a.off);
  a.off += ((n * sizeof(int) + 255) & ~(size_t)255);
  return p;
}
static inline dim3 g1(long n)  { return dim3((unsigned)((n + 255) / 256)); }
static inline dim3 gb(long n)  { return dim3((unsigned)((n + 255) / 256), BATCH); }

// lin: transpose weight [K,N] -> [N,K] into scratch, then GEMM (B transposed)
static inline void lin(hipStream_t st, Arena ar, const float* A, const float* W,
                       const float* bias, float* C, long rows, int N, int K, int act)
{
  float* wt = af(ar, (size_t)N * K);
  k_transpose_b<<<dim3((unsigned)(((long)K * N + 255) / 256), 1), 256, 0, st>>>(W, wt, K, N);
  int M = (int)rows;
  int tiles = ((M + 31) / 32) * ((N + 31) / 32);
  dim3 grid((tiles + NWAVES - 1) / NWAVES, 1);
  k_gemm_wmma<<<grid, 256, 0, st>>>(A, wt, bias, C, M, N, K, K, K, N, act, 1.f,
                                    1, 0, 0, 0, 0, 0, 0);
}

// Edge conv: fps(optional) -> knn(16) -> edge features -> 3x (GEMM+relu) -> max over K
static void edge_conv(hipStream_t st, Arena ar, const Lin* L, const int* od,
                      const float* feat, const float* xyz, int Nref, int Cin,
                      int out_np, float* out_feat, float* out_xyz)
{
  const int Kk = 16;
  const int M = out_np > 0 ? out_np : Nref;
  const float* qf; const float* qp;
  if (out_np > 0) {
    int* fidx = ai(ar, (size_t)BATCH * M);
    k_fps<<<BATCH, 256, 0, st>>>(xyz, fidx, Nref, M);
    float* qfb = af(ar, (size_t)BATCH * M * Cin);
    k_gather<<<gb((long)M * Cin), 256, 0, st>>>(feat, fidx, qfb, Nref, M, Cin);
    k_gather<<<gb((long)M * 3),   256, 0, st>>>(xyz,  fidx, out_xyz, Nref, M, 3);
    qf = qfb; qp = out_xyz;
  } else { qf = feat; qp = xyz; }
  int* nbr = ai(ar, (size_t)BATCH * M * Kk);
  k_knn<<<gb(M), 256, 0, st>>>(qp, xyz, nbr, nullptr, M, Nref, Kk);
  const long rows = (long)BATCH * M * Kk;
  int maxc = 2 * Cin;
  for (int i = 0; i < 3; ++i) if (od[i] > maxc) maxc = od[i];
  float* e0 = af(ar, (size_t)rows * maxc);
  float* e1 = af(ar, (size_t)rows * maxc);
  k_edge_feat<<<gb((long)M * Kk * Cin), 256, 0, st>>>(qf, feat, nbr, e0, M, Nref, Cin, Kk);
  const float* src = e0; float* dst = e1;
  int cin = 2 * Cin;
  for (int i = 0; i < 3; ++i) {
    lin(st, ar, src, L[i].w, L[i].b, dst, rows, od[i], cin, 1);
    const float* t = src; src = dst; dst = (float*)t;
    cin = od[i];
  }
  k_maxk<<<gb((long)M * cin), 256, 0, st>>>(src, out_feat, M, Kk, cin);
}

// Multi-head decode-transformer (H=8, centers Mc=1024)
static void dec_tr(hipStream_t st, Arena ar, const DT& p, const float* center,
                   const float* feat, const float* glob, int N, int C, float* out)
{
  const int Mc = 1024, H = 8, d = C / H;
  float* guide = af(ar, (size_t)BATCH * C);
  lin(st, ar, glob, p.guide.w, p.guide.b, guide, BATCH, C, 1024, 0);
  float* qin = af(ar, (size_t)BATCH * N * C);
  k_badd<<<gb((long)N * C), 256, 0, st>>>(feat, guide, qin, N, C);
  float* q = af(ar, (size_t)BATCH * N * C);
  lin(st, ar, qin, p.q.w, nullptr, q, (long)BATCH * N, C, C, 0);
  float* kvb = af(ar, (size_t)BATCH * Mc * C);
  lin(st, ar, center, p.pos.w, p.pos.b, kvb, (long)BATCH * Mc, C, 3, 0);
  float* kf = af(ar, (size_t)BATCH * Mc * C);
  lin(st, ar, kvb, p.k.w, nullptr, kf, (long)BATCH * Mc, C, C, 0);
  float* vf = af(ar, (size_t)BATCH * Mc * C);
  lin(st, ar, kvb, p.v.w, nullptr, vf, (long)BATCH * Mc, C, C, 0);
  // scores = softmax( Q @ K^T / sqrt(d) ) : kf is [b][m][c] -> B^T access
  // directly (bcol = kf + m*C + h*d, contiguous over k in the head slice).
  float* sc = af(ar, (size_t)BATCH * H * N * Mc);
  {
    int tiles = ((N + 31) / 32) * ((Mc + 31) / 32);
    dim3 grid((tiles + NWAVES - 1) / NWAVES, BATCH * H);
    k_gemm_wmma<<<grid, 256, 0, st>>>(q, kf, nullptr, sc, N, Mc, d, C, C, Mc,
        0, 1.f / sqrtf((float)d), H,
        (long)N * C, d, (long)Mc * C, d, (long)H * N * Mc, (long)N * Mc);
  }
  k_softmax<<<BATCH * H * N, 256, 0, st>>>(sc, Mc);
  // out = A @ V : pre-transpose V per batch -> vfT [b][C][Mc] so the B
  // operand is contiguous along k (= center index).
  float* vfT = af(ar, (size_t)BATCH * Mc * C);
  k_transpose_b<<<dim3((unsigned)(((long)Mc * C + 255) / 256), BATCH), 256, 0, st>>>(vf, vfT, Mc, C);
  float* o = af(ar, (size_t)BATCH * N * C);
  {
    int tiles = ((N + 31) / 32) * ((d + 31) / 32);
    dim3 grid((tiles + NWAVES - 1) / NWAVES, BATCH * H);
    k_gemm_wmma<<<grid, 256, 0, st>>>(sc, vfT, nullptr, o, N, d, Mc, Mc, Mc, C,
        0, 1.f, H,
        (long)H * N * Mc, (long)N * Mc, (long)Mc * C, (long)d * Mc, (long)N * C, d);
  }
  lin(st, ar, o, p.proj.w, p.proj.b, out, (long)BATCH * N, C, C, 0);
}

// Feature propagation: 3-NN inverse-distance interp + concat + 2x (GEMM+relu)
static void fp_up(hipStream_t st, Arena ar, const Lin* mlp, int n1, int n2,
                  const float* xyz1, int M, const float* xyz2, int Nsrc,
                  const float* f1, int C1, const float* f2, int C2, float* out)
{
  int* idx   = ai(ar, (size_t)BATCH * M * 3);
  float* dst = af(ar, (size_t)BATCH * M * 3);
  k_knn<<<gb(M), 256, 0, st>>>(xyz1, xyz2, idx, dst, M, Nsrc, 3);
  const int F = C1 + C2;
  float* h = af(ar, (size_t)BATCH * M * F);
  k_copy_cols<<<g1((long)BATCH * M * C1), 256, 0, st>>>(f1, C1, h, F, 0, (long)BATCH * M, C1);
  k_interp3<<<gb((long)M * C2), 256, 0, st>>>(f2, idx, dst, h, F, C1, M, Nsrc, C2);
  float* t1 = af(ar, (size_t)BATCH * M * n1);
  lin(st, ar, h,  mlp[0].w, mlp[0].b, t1,  (long)BATCH * M, n1, F,  1);
  lin(st, ar, t1, mlp[1].w, mlp[1].b, out, (long)BATCH * M, n2, n1, 1);
}

// Channel-attention feature fusion
static void caff_f(hipStream_t st, Arena ar, Lin l0, Lin l1,
                   const float* a, const float* bt, float* out, int N, int C)
{
  float* s = af(ar, (size_t)BATCH * C);
  k_colmean2<<<gb(C), 256, 0, st>>>(a, bt, s, N, C);
  float* hh = af(ar, (size_t)BATCH * (C / 4));
  lin(st, ar, s, l0.w, l0.b, hh, BATCH, C / 4, C, 1);
  float* g = af(ar, (size_t)BATCH * C);
  lin(st, ar, hh, l1.w, l1.b, g, BATCH, C, C / 4, 3);
  k_blend<<<gb((long)N * C), 256, 0, st>>>(g, a, bt, out, N, C);
}

// rebuild: concat(glob, head, c) -> rd -> FoldingNet (two 3-layer MLPs) -> output points
static void rebuild(hipStream_t st, Arena ar, const float* glob, const float* head, int Ch,
                    const float* cpts, int n, Lin rd, const Lin* f1, const Lin* f2,
                    float* outp, long ostride, long ooff)
{
  const int F = 1024 + Ch + 3;
  float* fbuf = af(ar, (size_t)BATCH * n * F);
  k_bcast_glob<<<gb((long)n * 1024), 256, 0, st>>>(glob, fbuf, F, 0, n, 1024);
  k_copy_cols<<<g1((long)BATCH * n * Ch), 256, 0, st>>>(head, Ch, fbuf, F, 1024, (long)BATCH * n, Ch);
  k_copy_cols<<<g1((long)BATCH * n * 3),  256, 0, st>>>(cpts, 3,  fbuf, F, 1024 + Ch, (long)BATCH * n, 3);
  float* xr = af(ar, (size_t)BATCH * n * 128);
  lin(st, ar, fbuf, rd.w, rd.b, xr, (long)BATCH * n, 128, F, 0);
  const long Mtot = (long)BATCH * n;
  const long rows = Mtot * 16;
  float* in1 = af(ar, (size_t)rows * 130);
  k_fold_in1<<<g1(rows * 130), 256, 0, st>>>(xr, in1, Mtot);
  float* t1  = af(ar, (size_t)rows * 128);
  float* t2  = af(ar, (size_t)rows * 64);
  float* fd1 = af(ar, (size_t)rows * 3);
  lin(st, ar, in1, f1[0].w, f1[0].b, t1,  rows, 128, 130, 1);
  lin(st, ar, t1,  f1[1].w, f1[1].b, t2,  rows, 64,  128, 1);
  lin(st, ar, t2,  f1[2].w, f1[2].b, fd1, rows, 3,   64,  0);
  float* in2 = af(ar, (size_t)rows * 131);
  k_fold_in2<<<g1(rows * 131), 256, 0, st>>>(xr, fd1, in2, Mtot);
  float* fd2 = af(ar, (size_t)rows * 3);
  lin(st, ar, in2, f2[0].w, f2[0].b, t1,  rows, 128, 131, 1);
  lin(st, ar, t1,  f2[1].w, f2[1].b, t2,  rows, 64,  128, 1);
  lin(st, ar, t2,  f2[2].w, f2[2].b, fd2, rows, 3,   64,  0);
  k_fold_out<<<gb((long)n * 48), 256, 0, st>>>(fd2, cpts, outp, n, ostride, ooff);
}

// ------------------------------- launch ------------------------------------

extern "C" void kernel_launch(void* const* d_in, const int* in_sizes, int n_in,
                              void* d_out, int out_size, void* d_ws, size_t ws_size,
                              hipStream_t stream)
{
  (void)in_sizes; (void)n_in; (void)out_size;
  hipStream_t st = stream;
  Arena ar; ar.base = (char*)d_ws; ar.off = 0; ar.cap = ws_size;

  // ---- unpack params in setup_inputs() dict order ----
  int cur = 0;
  const float* ptc = (const float*)d_in[cur++];            // [8,2048,3]
  auto nextL = [&](bool bias) {
    Lin l; l.w = (const float*)d_in[cur++];
    l.b = bias ? (const float*)d_in[cur++] : nullptr; return l;
  };
  auto next3 = [&](Lin* L) { for (int i = 0; i < 3; ++i) L[i] = nextL(true); };
  auto nextDT = [&]() {
    DT p; p.pos = nextL(true); p.guide = nextL(true);
    p.q = nextL(false); p.k = nextL(false); p.v = nextL(false);
    p.proj = nextL(true); return p;
  };
  Lin input_trans = nextL(true);
  Lin er1[3], er2[3], ers[3], ee1[3], ee2[3], ee3[3], ee4[3];
  next3(er1); next3(er2); next3(ers); next3(ee1); next3(ee2); next3(ee3); next3(ee4);
  Lin coarse1 = nextL(true), coarse2 = nextL(true);
  DT dectrans = nextDT();
  Lin conv1 = nextL(true);
  Lin up32_fp[2]; up32_fp[0] = nextL(true); up32_fp[1] = nextL(true);
  DT up32_tr = nextDT();
  Lin conv2 = nextL(true);
  Lin caff1[2]; caff1[0] = nextL(true); caff1[1] = nextL(true);
  Lin conv3 = nextL(true);
  Lin up21_fp[2]; up21_fp[0] = nextL(true); up21_fp[1] = nextL(true);
  DT up21_tr = nextDT();
  Lin conv4 = nextL(true);
  Lin caff2[2]; caff2[0] = nextL(true); caff2[1] = nextL(true);
  Lin conv5 = nextL(true);
  Lin down12[3]; next3(down12);
  Lin caff3[2]; caff3[0] = nextL(true); caff3[1] = nextL(true);
  Lin conv6 = nextL(true);
  Lin down23[3]; next3(down23);
  Lin caff4[2]; caff4[0] = nextL(true); caff4[1] = nextL(true);
  Lin rd1 = nextL(true), rd2 = nextL(true), rd3 = nextL(true);
  Lin fold1a[3], fold1b[3], fold2a[3], fold2b[3], fold3a[3], fold3b[3];
  next3(fold1a); next3(fold1b); next3(fold2a); next3(fold2b); next3(fold3a); next3(fold3b);

  // ---- forward ----
  float* x = af(ar, (size_t)BATCH * 2048 * 8);
  lin(st, ar, ptc, input_trans.w, input_trans.b, x, (long)BATCH * 2048, 8, 3, 0);

  // EdgeRes
  const int od_er1[3] = {8, 8, 16};
  float* hf = af(ar, (size_t)BATCH * 1536 * 16);
  float* hp = af(ar, (size_t)BATCH * 1536 * 3);
  edge_conv(st, ar, er1, od_er1, x, ptc, 2048, 8, 1536, hf, hp);
  const int od_er2[3] = {16, 16, 32};
  float* o1f = af(ar, (size_t)BATCH * 1024 * 32);
  float* xp1 = af(ar, (size_t)BATCH * 1024 * 3);   // discarded positions
  edge_conv(st, ar, er2, od_er2, hf, hp, 1536, 16, 1024, o1f, xp1);
  const int od_ers[3] = {8, 16, 32};
  float* o2f = af(ar, (size_t)BATCH * 1024 * 32);
  float* op  = af(ar, (size_t)BATCH * 1024 * 3);
  edge_conv(st, ar, ers, od_ers, x, ptc, 2048, 8, 1024, o2f, op);
  float* ef = af(ar, (size_t)BATCH * 1024 * 32);
  k_add<<<g1((long)BATCH * 1024 * 32), 256, 0, st>>>(o1f, o2f, ef, (long)BATCH * 1024 * 32);

  // EdgeExtractFeature pyramid
  const int od_ee1[3] = {64, 64, 128};
  float* s1f = af(ar, (size_t)BATCH * 512 * 128);
  float* s1p = af(ar, (size_t)BATCH * 512 * 3);
  edge_conv(st, ar, ee1, od_ee1, ef, op, 1024, 32, 512, s1f, s1p);
  const int od_ee2[3] = {128, 128, 256};
  float* s2f = af(ar, (size_t)BATCH * 256 * 256);
  float* s2p = af(ar, (size_t)BATCH * 256 * 3);
  edge_conv(st, ar, ee2, od_ee2, s1f, s1p, 512, 128, 256, s2f, s2p);
  const int od_ee3[3] = {256, 256, 512};
  float* s3f = af(ar, (size_t)BATCH * 128 * 512);
  float* s3p = af(ar, (size_t)BATCH * 128 * 3);
  edge_conv(st, ar, ee3, od_ee3, s2f, s2p, 256, 256, 128, s3f, s3p);
  const int od_ee4[3] = {512, 512, 1024};
  float* gfb = af(ar, (size_t)BATCH * 128 * 1024);
  edge_conv(st, ar, ee4, od_ee4, s3f, s3p, 128, 512, 0, gfb, nullptr);
  float* glob = af(ar, (size_t)BATCH * 1024);
  k_colmax<<<gb(1024), 256, 0, st>>>(gfb, glob, 128, 1024);

  // coarse centers
  float* chh = af(ar, (size_t)BATCH * 1024);
  lin(st, ar, glob, coarse1.w, coarse1.b, chh, BATCH, 1024, 1024, 1);
  float* craw = af(ar, (size_t)BATCH * 3072);
  lin(st, ar, chh, coarse2.w, coarse2.b, craw, BATCH, 3072, 1024, 0);
  float* center = af(ar, (size_t)BATCH * 1024 * 3);
  k_center<<<gb(1024 * 3), 256, 0, st>>>(craw, center, (float*)d_out, 1024);

  // decode / upsample / fuse
  float* dec = af(ar, (size_t)BATCH * 128 * 512);
  dec_tr(st, ar, dectrans, center, s3f, glob, 128, 512, dec);
  float* s3 = af(ar, (size_t)BATCH * 128 * 512);
  lin(st, ar, dec, conv1.w, conv1.b, s3, (long)BATCH * 128, 512, 512, 2);

  float* fp32o = af(ar, (size_t)BATCH * 256 * 256);
  fp_up(st, ar, up32_fp, 256, 256, s2p, 256, s3p, 128, s2f, 256, s3, 512, fp32o);
  float* s32 = af(ar, (size_t)BATCH * 256 * 256);
  dec_tr(st, ar, up32_tr, center, fp32o, glob, 256, 256, s32);
  float* s2 = af(ar, (size_t)BATCH * 256 * 256);
  lin(st, ar, s2f, conv2.w, conv2.b, s2, (long)BATCH * 256, 256, 256, 2);
  float* cf1 = af(ar, (size_t)BATCH * 256 * 256);
  caff_f(st, ar, caff1[0], caff1[1], s32, s2, cf1, 256, 256);
  float* s2f1 = af(ar, (size_t)BATCH * 256 * 256);
  lin(st, ar, cf1, conv3.w, conv3.b, s2f1, (long)BATCH * 256, 256, 256, 2);

  float* fp21o = af(ar, (size_t)BATCH * 512 * 128);
  fp_up(st, ar, up21_fp, 128, 128, s1p, 512, s2p, 256, s1f, 128, s2f1, 256, fp21o);
  float* s21 = af(ar, (size_t)BATCH * 512 * 128);
  dec_tr(st, ar, up21_tr, center, fp21o, glob, 512, 128, s21);
  float* s1 = af(ar, (size_t)BATCH * 512 * 128);
  lin(st, ar, s1f, conv4.w, conv4.b, s1, (long)BATCH * 512, 128, 128, 2);
  float* cf2 = af(ar, (size_t)BATCH * 512 * 128);
  caff_f(st, ar, caff2[0], caff2[1], s21, s1, cf2, 512, 128);
  float* s1fu = af(ar, (size_t)BATCH * 512 * 128);
  lin(st, ar, cf2, conv5.w, conv5.b, s1fu, (long)BATCH * 512, 128, 128, 2);

  const int od_d12[3] = {128, 128, 256};
  float* s12 = af(ar, (size_t)BATCH * 256 * 256);
  float* xp2 = af(ar, (size_t)BATCH * 256 * 3);
  edge_conv(st, ar, down12, od_d12, s1fu, s1p, 512, 128, 256, s12, xp2);
  float* cf3 = af(ar, (size_t)BATCH * 256 * 256);
  caff_f(st, ar, caff3[0], caff3[1], s12, s2f1, cf3, 256, 256);
  float* s2f2 = af(ar, (size_t)BATCH * 256 * 256);
  lin(st, ar, cf3, conv6.w, conv6.b, s2f2, (long)BATCH * 256, 256, 256, 2);

  const int od_d23[3] = {256, 256, 512};
  float* s23 = af(ar, (size_t)BATCH * 128 * 512);
  float* xp3 = af(ar, (size_t)BATCH * 128 * 3);
  edge_conv(st, ar, down23, od_d23, s2f2, s2p, 256, 256, 128, s23, xp3);
  float* s3fu = af(ar, (size_t)BATCH * 128 * 512);
  caff_f(st, ar, caff4[0], caff4[1], s23, s3, s3fu, 128, 512);

  // FPS on centers for the three rebuild levels
  int* ci1 = ai(ar, (size_t)BATCH * 512);
  int* ci2 = ai(ar, (size_t)BATCH * 256);
  int* ci3 = ai(ar, (size_t)BATCH * 128);
  k_fps<<<BATCH, 256, 0, st>>>(center, ci1, 1024, 512);
  k_fps<<<BATCH, 256, 0, st>>>(center, ci2, 1024, 256);
  k_fps<<<BATCH, 256, 0, st>>>(center, ci3, 1024, 128);
  float* c1 = af(ar, (size_t)BATCH * 512 * 3);
  float* c2 = af(ar, (size_t)BATCH * 256 * 3);
  float* c3 = af(ar, (size_t)BATCH * 128 * 3);
  k_gather<<<gb(512L * 3), 256, 0, st>>>(center, ci1, c1, 1024, 512, 3);
  k_gather<<<gb(256L * 3), 256, 0, st>>>(center, ci2, c2, 1024, 256, 3);
  k_gather<<<gb(128L * 3), 256, 0, st>>>(center, ci3, c3, 1024, 128, 3);

  // rebuild into d_out: [center | concat(r1 (8192), r2 (4096), r3 (2048)) per batch]
  float* out2 = (float*)d_out + (long)BATCH * 1024 * 3;
  const long ostride = 14336L * 3;
  rebuild(st, ar, glob, s1fu, 128, c1, 512, rd1, fold1a, fold1b, out2, ostride, 0L);
  rebuild(st, ar, glob, s2f2, 256, c2, 256, rd2, fold2a, fold2b, out2, ostride, 8192L * 3);
  rebuild(st, ar, glob, s3fu, 512, c3, 128, rd3, fold3a, fold3b, out2, ostride, 12288L * 3);
}
